// AdaptiveDiffusionLayer_78288663872054
// MI455X (gfx1250) — matrix operand order repaired and checked by reading, not compiled, not run on hardware
//
#include <hip/hip_runtime.h>

typedef __attribute__((ext_vector_type(16))) _Float16 v16h;
typedef __attribute__((ext_vector_type(8)))  _Float16 v8h;
typedef __attribute__((ext_vector_type(4)))  _Float16 v4h;
typedef __attribute__((ext_vector_type(8)))  float    v8f;
typedef __attribute__((ext_vector_type(4)))  float    v4f;

#define BM      64
#define BN      512
#define BK      32
#define APAD    40   // halves per A row (80B, multiple of 16B)
#define BPAD    40   // halves per B column-row (transposed [n][k])
#define NSUB    8    // 16x16 subtiles per wave (wave tile 16x128)
#define THREADS 512  // 16 wave32s

// C[M x 512] = A[M x K] (lda) @ B[K x 512], f16 WMMA, f32 accumulate.
// blend=1: C = (1-t)*blendsrc + t*acc
__global__ __launch_bounds__(THREADS) void gcn_gemm_wmma_f16(
    const float* __restrict__ A, const float* __restrict__ B,
    float* __restrict__ C, int M, int K, long lda,
    const float* __restrict__ t, const float* __restrict__ blendsrc,
    int blend, int nt_a)
{
    // double-buffered staging: 2 * (5KB + 40KB) = 90KB of the 320KB WGP LDS
    __shared__ __align__(16) _Float16 sA[2][BM * APAD];
    __shared__ __align__(16) _Float16 sB[2][BN * BPAD];

    const int tid  = threadIdx.x;
    const int lane = tid & 31;
    const int wave = tid >> 5;      // 0..15
    const int wm   = wave >> 2;     // 0..3 : M sub-row (16 rows each)
    const int wn   = wave & 3;      // 0..3 : N quarter (128 wide)
    const int row0 = blockIdx.x * BM;

    // ---- staging decomposition (fixed per thread) ----
    // A tile: 64x32 = 512 float4, 1 per thread (k-contiguous row-major)
    const int a_m = tid >> 3;         // 0..63
    const int a_k = (tid & 7) * 4;    // 0..28
    // B tile: 32x512 in 4k x 4n units; 1024 units, 2 per thread

    v8f acc[NSUB];
#pragma unroll
    for (int s = 0; s < NSUB; ++s)
#pragma unroll
        for (int i = 0; i < 8; ++i) acc[s][i] = 0.0f;

    // in-flight staging registers (one tile ahead of LDS)
    v4f aval;
    v4f bval[2][4];   // [unit][dk]

    auto load_tile = [&](int ks) {
        const int k0 = ks * BK;
        {   // A (adj/x) — optionally non-temporal stream
            const int grow = row0 + a_m;
            const int gk   = k0 + a_k;
            v4f v = {0.f, 0.f, 0.f, 0.f};
            if (grow < M && gk < K) {   // K%4==0 -> float4 fully in/out
                const v4f* p = (const v4f*)(A + (long)grow * lda + gk);
                v = nt_a ? __builtin_nontemporal_load(p) : *p;
            }
            aval = v;
        }
#pragma unroll
        for (int u = 0; u < 2; ++u) {   // B (support/W) — L2 resident
            const int uu = tid + u * THREADS;
            const int kb = uu >> 7;
            const int n  = (uu & 127) * 4;
#pragma unroll
            for (int dk = 0; dk < 4; ++dk) {
                const int gk = k0 + kb * 4 + dk;
                v4f v = {0.f, 0.f, 0.f, 0.f};
                if (gk < K) v = *(const v4f*)(B + (long)gk * 512 + n);
                bval[u][dk] = v;
            }
        }
    };

    auto store_tile = [&](int buf) {
        {   // A: 4 k-contiguous halves -> one ds_store_b64
            v4h h;
            h[0] = (_Float16)aval.x; h[1] = (_Float16)aval.y;
            h[2] = (_Float16)aval.z; h[3] = (_Float16)aval.w;
            *(v4h*)&sA[buf][a_m * APAD + a_k] = h;
        }
#pragma unroll
        for (int u = 0; u < 2; ++u) {   // B: transpose 4k x 4n in regs, b64 stores
            const int uu = tid + u * THREADS;
            const int kb = uu >> 7;
            const int n  = (uu & 127) * 4;
#pragma unroll
            for (int e = 0; e < 4; ++e) {
                v4h h;
                h[0] = (_Float16)bval[u][0][e]; h[1] = (_Float16)bval[u][1][e];
                h[2] = (_Float16)bval[u][2][e]; h[3] = (_Float16)bval[u][3][e];
                *(v4h*)&sB[buf][(n + e) * BPAD + kb * 4] = h;
            }
        }
    };

    // fragment addressing (documented 16-bit WMMA layouts)
    const int arow  = wm * 16 + (lane & 15);
    const int abase = (lane >> 4) * 8;     // A: lanes>=16 hold K {8..15,24..31}
    const int bn_l  = lane & 15;
    const int bbase = (lane >> 4) * 8;     // B: lanes>=16 hold K {8..15,24..31}

    auto compute = [&](int buf) {
        v8h a_lo = *(const v8h*)&sA[buf][arow * APAD + abase];
        v8h a_hi = *(const v8h*)&sA[buf][arow * APAD + abase + 16];
        v16h af;
#pragma unroll
        for (int i = 0; i < 8; ++i) { af[i] = a_lo[i]; af[i + 8] = a_hi[i]; }

        // B-fragment register double buffer: issue loads for subtile s+1
        // before the WMMA for subtile s so DS latency overlaps the matrix op.
        const _Float16* bp0 = &sB[buf][(wn * 128 + bn_l) * BPAD + bbase];
        v8h b_lo = *(const v8h*)(bp0);
        v8h b_hi = *(const v8h*)(bp0 + 16);
#pragma unroll
        for (int s = 0; s < NSUB; ++s) {
            v16h bf;
#pragma unroll
            for (int i = 0; i < 8; ++i) { bf[i] = b_lo[i]; bf[i + 8] = b_hi[i]; }
            if (s + 1 < NSUB) {
                const _Float16* bp = bp0 + (s + 1) * 16 * BPAD;
                b_lo = *(const v8h*)(bp);
                b_hi = *(const v8h*)(bp + 16);
            }
            acc[s] = __builtin_amdgcn_wmma_f32_16x16x32_f16(
                false, af, false, bf, (short)0, acc[s], false, false);
        }
    };

    // ---- 2-deep software pipeline, one barrier per k-step ----
    const int ksteps = (K + BK - 1) / BK;
    load_tile(0);
    store_tile(0);
    load_tile(1);                     // zero-fills harmlessly if ksteps==1
    for (int ks = 0; ks < ksteps; ++ks) {
        __syncthreads();              // buf(cur) staged; buf(cur^1) drained
        const int cur = ks & 1;
        if (ks + 1 < ksteps) store_tile(cur ^ 1);   // tile ks+1 regs -> LDS
        if (ks + 2 < ksteps) load_tile(ks + 2);     // issue HBM loads early
        compute(cur);
    }

    // ---- epilogue: D lane L: N = L&15; VGPR r -> M = r + 8*(L>=16) ----
    const float ts       = blend ? t[0] : 0.0f;
    const int   rbase    = row0 + wm * 16 + (lane >> 4) * 8;
    const int   col_base = wn * 128 + (lane & 15);
#pragma unroll
    for (int s = 0; s < NSUB; ++s) {
        const int col = col_base + s * 16;
#pragma unroll
        for (int r = 0; r < 8; ++r) {
            const int row = rbase + r;
            if (row < M) {
                float v = acc[s][r];
                if (blend) {
                    const float sup = blendsrc[(long)row * 512 + col];
                    v = (1.0f - ts) * sup + ts * v;
                }
                C[(long)row * 512 + col] = v;
            }
        }
    }
}

extern "C" void kernel_launch(void* const* d_in, const int* in_sizes, int n_in,
                              void* d_out, int out_size, void* d_ws, size_t ws_size,
                              hipStream_t stream) {
    (void)in_sizes; (void)n_in; (void)out_size; (void)ws_size;
    const float* x      = (const float*)d_in[0];   // [10000, 512]
    const float* adj    = (const float*)d_in[1];   // [10000, 10000]
    const float* t      = (const float*)d_in[2];   // [1]
    const float* weight = (const float*)d_in[3];   // [512, 512]
    float* out     = (float*)d_out;                // [10000, 512]
    float* support = (float*)d_ws;                 // [10000, 512] fp32 scratch

    const int M = 10000;
    dim3 block(THREADS);
    dim3 grid((M + BM - 1) / BM);                  // 157 blocks

    // Pass 1: support = x @ W   (K = 512)
    gcn_gemm_wmma_f16<<<grid, block, 0, stream>>>(
        x, weight, support, M, 512, 512L, t, nullptr, 0, 0);

    // Pass 2: out = (1-t)*support + t*(adj @ support)
    //         adj streamed NT (one-shot 400MB) so support stays L2-resident
    gcn_gemm_wmma_f16<<<grid, block, 0, stream>>>(
        adj, support, out, M, 10000, 10000L, t, support, 1, 1);
}
